// NiceModel_35923106463835
// MI455X (gfx1250) — compile-verified, matrix operand order checked by reading, not run on hardware
//
#include <hip/hip_runtime.h>

// ---------------------------------------------------------------------------
// GNN forward (GCN + 3xGAT + GCN + pairwise rank head) for MI455X / gfx1250.
// Node GEMMs via v_wmma_f32_16x16x32_f16 (wave32, 16x16 tile per wave).
// Rank head factored: z = u_i + v_j + b1 ; LN stats in closed form; fused
// GELU-dot epilogue (no 260096x256 matrix ever materialized).
// ---------------------------------------------------------------------------

typedef __attribute__((ext_vector_type(16))) _Float16 v16h;
typedef __attribute__((ext_vector_type(8)))  _Float16 v8h;
typedef __attribute__((ext_vector_type(8)))  float    v8f;

#define EPSV 1e-5f
#define LRS  0.2f

__device__ __forceinline__ float lrelu(float x) { return x > 0.f ? x : LRS * x; }
// order-preserving float -> uint key (for atomicMax-based segment max)
__device__ __forceinline__ unsigned fkey(float f) {
    unsigned b = __float_as_uint(f);
    return (b & 0x80000000u) ? ~b : (b | 0x80000000u);
}
__device__ __forceinline__ float funkey(unsigned k) {
    return __uint_as_float((k & 0x80000000u) ? (k & 0x7FFFFFFFu) : ~k);
}

// ----------------------------- tiny utilities ------------------------------
__global__ void k_fill(float* p, float v, int n) {
    int i = blockIdx.x * blockDim.x + threadIdx.x;
    if (i < n) p[i] = v;
}
__global__ void k_count_deg(const int* __restrict__ dst, float* deg, int E) {
    int i = blockIdx.x * blockDim.x + threadIdx.x;
    if (i < E) atomicAdd(&deg[dst[i]], 1.0f);
}
__global__ void k_rsqrt_inplace(float* p, int n) {
    int i = blockIdx.x * blockDim.x + threadIdx.x;
    if (i < n) p[i] = rsqrtf(p[i]);
}
__global__ void k_f32_to_f16(const float* __restrict__ in, _Float16* __restrict__ out, int n) {
    int i = blockIdx.x * blockDim.x + threadIdx.x;
    if (i < n) out[i] = (_Float16)in[i];
}
// in is K x N (row-major f32) -> out is N x K (row-major f16)
__global__ void k_transpose_to_f16(const float* __restrict__ in, _Float16* __restrict__ out,
                                   int K, int N) {
    int i = blockIdx.x * blockDim.x + threadIdx.x;
    if (i >= K * N) return;
    int k = i / N, n = i % N;
    out[(size_t)n * K + k] = (_Float16)in[i];
}
__global__ void k_add(const float* __restrict__ a, const float* __restrict__ b,
                      float* __restrict__ o, int n) {
    int i = blockIdx.x * blockDim.x + threadIdx.x;
    if (i < n) o[i] = a[i] + b[i];
}

// ----------------------------- WMMA GEMM -----------------------------------
// C[M x N] = A[M x K] * B  with B given transposed as Bt[N x K]; A,Bt f16,
// C f32; optional bias (length N). One wave -> one 16x16 tile.
__global__ void k_wmma_gemm(const _Float16* __restrict__ A, const _Float16* __restrict__ Bt,
                            const float* __restrict__ bias, float* __restrict__ C,
                            int M, int N, int K) {
    int wid  = (int)((blockIdx.x * (long)blockDim.x + threadIdx.x) >> 5);
    int lane = threadIdx.x & 31;
    int tilesN = N >> 4;
    int tM = (wid / tilesN) << 4;
    int tN = (wid % tilesN) << 4;
    if (tM >= M) return;
    int half = lane >> 4;   // lane group
    int r    = lane & 15;   // row (A) / col (B,D)

    const _Float16* arow = A  + (size_t)(tM + r) * K + 8 * half;
    const _Float16* brow = Bt + (size_t)(tN + r) * K + 8 * half;

    v8f acc = {0.f, 0.f, 0.f, 0.f, 0.f, 0.f, 0.f, 0.f};
    for (int k0 = 0; k0 < K; k0 += 32) {
        v8h a0 = *reinterpret_cast<const v8h*>(arow + k0);
        v8h a1 = *reinterpret_cast<const v8h*>(arow + k0 + 16);
        v8h b0 = *reinterpret_cast<const v8h*>(brow + k0);
        v8h b1 = *reinterpret_cast<const v8h*>(brow + k0 + 16);
        v16h af, bf;
#pragma unroll
        for (int t = 0; t < 8; ++t) {
            af[t] = a0[t]; af[t + 8] = a1[t];
            bf[t] = b0[t]; bf[t + 8] = b1[t];
        }
        acc = __builtin_amdgcn_wmma_f32_16x16x32_f16(false, af, false, bf,
                                                     (short)0, acc, false, false);
    }
    float bv = bias ? bias[tN + r] : 0.f;
    float* cp = C + (size_t)tM * N + tN + r;
#pragma unroll
    for (int t = 0; t < 8; ++t) {
        int m = t + 8 * half;                 // D layout: M = vgpr + 8*lanegrp
        cp[(size_t)m * N] = acc[t] + bv;      // N = lane & 15
    }
}

// ------------------------- graph aggregation -------------------------------
// out[v,f] = xin[v,f] * dinv[v]^2 (+ bias[f])   (self-loop term / init)
__global__ void k_self_norm(const float* __restrict__ xin, const float* __restrict__ dinv,
                            const float* __restrict__ bias, float* __restrict__ out,
                            int Nn, int F) {
    int i = blockIdx.x * blockDim.x + threadIdx.x;
    if (i >= Nn * F) return;
    int v = i / F, f = i % F;
    float dv = dinv[v];
    out[i] = xin[i] * dv * dv + (bias ? bias[f] : 0.f);
}
// out[dst] += xin[src] * dinv[src]*dinv[dst], float4 per thread
__global__ void k_edge_scatter_norm(const float* __restrict__ xin, const float* __restrict__ dinv,
                                    const int* __restrict__ src, const int* __restrict__ dst,
                                    float* __restrict__ out, int E, int F) {
    int F4 = F >> 2;
    long i = blockIdx.x * (long)blockDim.x + threadIdx.x;
    if (i >= (long)E * F4) return;
    int e = (int)(i / F4), c = (int)(i % F4);
    int s = src[e], d = dst[e];
    float nm = dinv[s] * dinv[d];
    float4 xv = reinterpret_cast<const float4*>(xin + (size_t)s * F)[c];
    float* op = out + (size_t)d * F + c * 4;
    atomicAdd(op + 0, xv.x * nm);
    atomicAdd(op + 1, xv.y * nm);
    atomicAdd(op + 2, xv.z * nm);
    atomicAdd(op + 3, xv.w * nm);
}

// ------------------------------ batch norm ---------------------------------
__global__ void k_bn_stats(const float* __restrict__ x, float* __restrict__ mean,
                           float* __restrict__ var, int Nn, int F) {
    __shared__ float s1[256];
    __shared__ float s2[256];
    int f = blockIdx.x, t = threadIdx.x;
    float a = 0.f, b = 0.f;
    for (int i = t; i < Nn; i += blockDim.x) {
        float v = x[(size_t)i * F + f];
        a += v; b += v * v;
    }
    s1[t] = a; s2[t] = b;
    __syncthreads();
    for (int o = 128; o > 0; o >>= 1) {
        if (t < o) { s1[t] += s1[t + o]; s2[t] += s2[t + o]; }
        __syncthreads();
    }
    if (t == 0) {
        float mu = s1[0] / Nn;
        mean[f] = mu;
        var[f]  = s2[0] / Nn - mu * mu;
    }
}
__global__ void k_bn_relu(const float* __restrict__ x, const float* __restrict__ mean,
                          const float* __restrict__ var, const float* __restrict__ g,
                          const float* __restrict__ b, float* __restrict__ out,
                          int Nn, int F) {
    int i = blockIdx.x * blockDim.x + threadIdx.x;
    if (i >= Nn * F) return;
    int f = i % F;
    float t = (x[i] - mean[f]) * rsqrtf(var[f] + EPSV) * g[f] + b[f];
    out[i] = fmaxf(t, 0.f);
}

// --------------------------------- GAT -------------------------------------
__global__ void k_gat_dots(const float* __restrict__ xw, const float* __restrict__ asrc,
                           const float* __restrict__ adst, float* __restrict__ als,
                           float* __restrict__ ald, int Nn, int F) {
    __shared__ float s1[256];
    __shared__ float s2[256];
    int v = blockIdx.x, t = threadIdx.x;
    float a = 0.f, b = 0.f;
    for (int f = t; f < F; f += blockDim.x) {
        float x = xw[(size_t)v * F + f];
        a += x * asrc[f]; b += x * adst[f];
    }
    s1[t] = a; s2[t] = b;
    __syncthreads();
    for (int o = 128; o > 0; o >>= 1) {
        if (t < o) { s1[t] += s1[t + o]; s2[t] += s2[t + o]; }
        __syncthreads();
    }
    if (t == 0) { als[v] = s1[0]; ald[v] = s2[0]; }
}
__global__ void k_init_maxsum(unsigned* nmax, float* nsum, int n) {
    int i = blockIdx.x * blockDim.x + threadIdx.x;
    if (i < n) { nmax[i] = 0u; nsum[i] = 0.f; }  // key 0 == -inf sentinel
}
__global__ void k_gat_max(const int* __restrict__ src, const int* __restrict__ dst,
                          const float* __restrict__ als, const float* __restrict__ ald,
                          unsigned* nmax, int E, int Nn) {
    int i = blockIdx.x * blockDim.x + threadIdx.x;
    if (i >= E + Nn) return;
    int s, d;
    if (i < E) { s = src[i]; d = dst[i]; } else { s = d = i - E; }  // self loops
    atomicMax(&nmax[d], fkey(lrelu(als[s] + ald[d])));
}
__global__ void k_gat_sum(const int* __restrict__ src, const int* __restrict__ dst,
                          const float* __restrict__ als, const float* __restrict__ ald,
                          const unsigned* __restrict__ nmax, float* nsum, int E, int Nn) {
    int i = blockIdx.x * blockDim.x + threadIdx.x;
    if (i >= E + Nn) return;
    int s, d;
    if (i < E) { s = src[i]; d = dst[i]; } else { s = d = i - E; }
    float e = lrelu(als[s] + ald[d]);
    atomicAdd(&nsum[d], expf(e - funkey(nmax[d])));
}
__global__ void k_gat_self_out(const float* __restrict__ xw, const float* __restrict__ als,
                               const float* __restrict__ ald, const unsigned* __restrict__ nmax,
                               const float* __restrict__ nsum, const float* __restrict__ bias,
                               float* __restrict__ out, int Nn, int F) {
    int i = blockIdx.x * blockDim.x + threadIdx.x;
    if (i >= Nn * F) return;
    int v = i / F, f = i % F;
    float e = lrelu(als[v] + ald[v]);
    float w = expf(e - funkey(nmax[v])) / nsum[v];
    out[i] = bias[f] + xw[i] * w;
}
__global__ void k_gat_edge_scatter(const float* __restrict__ xw, const int* __restrict__ src,
                                   const int* __restrict__ dst, const float* __restrict__ als,
                                   const float* __restrict__ ald, const unsigned* __restrict__ nmax,
                                   const float* __restrict__ nsum, float* __restrict__ out,
                                   int E, int F) {
    int F4 = F >> 2;
    long i = blockIdx.x * (long)blockDim.x + threadIdx.x;
    if (i >= (long)E * F4) return;
    int e = (int)(i / F4), c = (int)(i % F4);
    int s = src[e], d = dst[e];
    float a = expf(lrelu(als[s] + ald[d]) - funkey(nmax[d])) / nsum[d];
    float4 xv = reinterpret_cast<const float4*>(xw + (size_t)s * F)[c];
    float* op = out + (size_t)d * F + c * 4;
    atomicAdd(op + 0, xv.x * a);
    atomicAdd(op + 1, xv.y * a);
    atomicAdd(op + 2, xv.z * a);
    atomicAdd(op + 3, xv.w * a);
}

// ----------------------------- rank head -----------------------------------
// Closed-form global LN stats over z[g,i<j,k] = u[g,i,k] + v[g,j,k]  (v has b1)
__global__ void k_rank_stats(const float* __restrict__ u, const float* __restrict__ v,
                             float* __restrict__ S, int G, int P, int F) {
    int i = blockIdx.x * blockDim.x + threadIdx.x;
    if (i >= G * F) return;
    int g = i / F, k = i % F;
    const float* up = u + (size_t)g * P * F + k;
    const float* vp = v + (size_t)g * P * F + k;
    float s1 = 0.f, s2 = 0.f, pref = 0.f;
    for (int r = 0; r < P; ++r) {
        float uu = up[(size_t)r * F];
        float vv = vp[(size_t)r * F];
        float wu = (float)(P - 1 - r);
        float wv = (float)r;
        s1 += uu * wu + vv * wv;
        s2 += uu * uu * wu + vv * vv * wv + 2.f * vv * pref;
        pref += uu;
    }
    atomicAdd(&S[0], s1);
    atomicAdd(&S[1], s2);
}
__global__ void k_rank_ms(const float* __restrict__ S, float* __restrict__ ms, float T) {
    float mu = S[0] / T;
    float var = S[1] / T - mu * mu;
    float sd = sqrtf(fmaxf(var, 0.f));
    ms[0] = mu;
    ms[1] = 1.f / (sd + EPSV);
}
// One wave per (i<j) pair: fused LN + exact GELU + dot(W2)
__global__ void k_rank_out(const float* __restrict__ u, const float* __restrict__ v,
                           const float* __restrict__ ms, const float* __restrict__ ln_g,
                           const float* __restrict__ ln_b, const float* __restrict__ w2,
                           const float* __restrict__ b2, float* __restrict__ out,
                           int G, int P, int F, int PAIRS) {
    int wid  = (int)((blockIdx.x * (long)blockDim.x + threadIdx.x) >> 5);
    int lane = threadIdx.x & 31;
    if (wid >= G * PAIRS) return;
    int g = wid / PAIRS;
    int p = wid % PAIRS;
    // invert row-major triu(P,1) index p -> (i,j)
    float Pf = (float)P;
    float disc = (2.f * Pf - 1.f) * (2.f * Pf - 1.f) - 8.f * (float)p;
    int i = (int)floorf((2.f * Pf - 1.f - sqrtf(fmaxf(disc, 0.f))) * 0.5f);
    if (i < 0) i = 0;
    if (i > P - 2) i = P - 2;
    auto off = [P](int t) { return t * (P - 1) - (t * (t - 1)) / 2; };
    while (i > 0 && off(i) > p) --i;
    while (off(i + 1) <= p) ++i;
    int j = p - off(i) + i + 1;

    float mu = ms[0], sc = ms[1];
    const float* up = u + ((size_t)g * P + i) * F;
    const float* vp = v + ((size_t)g * P + j) * F;
    float acc = 0.f;
    for (int k = lane; k < F; k += 32) {
        float z = (up[k] + vp[k] - mu) * sc * ln_g[k] + ln_b[k];
        float ge = 0.5f * z * (1.f + erff(z * 0.70710678118f));  // exact GELU
        acc += ge * w2[k];
    }
    for (int o = 16; o > 0; o >>= 1) acc += __shfl_xor(acc, o, 32);
    if (lane == 0) out[wid] = acc + b2[0];
}

// ------------------------------- driver ------------------------------------
extern "C" void kernel_launch(void* const* d_in, const int* in_sizes, int n_in,
                              void* d_out, int out_size, void* d_ws, size_t ws_size,
                              hipStream_t stream) {
    const int N = 4096, E = 65536, FIN = 128, H = 256;
    const int G = 32, P = 128, PAIRS = P * (P - 1) / 2;  // 8128
    const float Ttot = (float)((double)G * PAIRS * H);

    const float* x       = (const float*)d_in[0];
    const int*   srcI    = (const int*)d_in[1];
    const int*   dstI    = srcI + E;
    const float* gcn1_W  = (const float*)d_in[3];
    const float* gcn1_b  = (const float*)d_in[4];
    const float* gcn2_W  = (const float*)d_in[5];
    const float* gcn2_b  = (const float*)d_in[6];
    const float* rank_W1 = (const float*)d_in[7];
    const float* rank_b1 = (const float*)d_in[8];
    const float* rank_W2 = (const float*)d_in[9];
    const float* rank_b2 = (const float*)d_in[10];
    const float* ln_g    = (const float*)d_in[11];
    const float* ln_b    = (const float*)d_in[12];
    const float* bn_g[4] = {(const float*)d_in[13], (const float*)d_in[15],
                            (const float*)d_in[17], (const float*)d_in[19]};
    const float* bn_b[4] = {(const float*)d_in[14], (const float*)d_in[16],
                            (const float*)d_in[18], (const float*)d_in[20]};
    const float *gat_W[3], *gat_as[3], *gat_ad[3], *gat_bb[3];
    for (int i = 0; i < 3; ++i) {
        gat_W[i]  = (const float*)d_in[21 + 4 * i];
        gat_as[i] = (const float*)d_in[22 + 4 * i];
        gat_ad[i] = (const float*)d_in[23 + 4 * i];
        gat_bb[i] = (const float*)d_in[24 + 4 * i];
    }

    // workspace carve-out
    char* wsp = (char*)d_ws;
    size_t oo = 0;
    auto carve = [&](size_t bytes) -> void* {
        void* p = wsp + oo;
        oo = (oo + bytes + 255) & ~(size_t)255;
        return p;
    };
    float*    dinv = (float*)carve((size_t)N * 4);
    float*    hbuf = (float*)carve((size_t)N * H * 4);
    float*    xw   = (float*)carve((size_t)N * H * 4);
    float*    agg  = (float*)carve((size_t)N * H * 4);
    float*    res  = (float*)carve((size_t)N * H * 4);
    float*    ub   = (float*)carve((size_t)N * H * 4);
    float*    vb   = (float*)carve((size_t)N * H * 4);
    _Float16* a16  = (_Float16*)carve((size_t)N * H * 2);
    _Float16* w16  = (_Float16*)carve((size_t)H * H * 2);
    float*    als  = (float*)carve((size_t)N * 4);
    float*    ald  = (float*)carve((size_t)N * 4);
    unsigned* nmax = (unsigned*)carve((size_t)N * 4);
    float*    nsum = (float*)carve((size_t)N * 4);
    float*    Sst  = (float*)carve(8);
    float*    ms   = (float*)carve(8);
    float*    bnm  = (float*)carve((size_t)H * 4);
    float*    bnv  = (float*)carve((size_t)H * 4);
    (void)ws_size; (void)n_in; (void)in_sizes; (void)out_size;

    const int TB = 256;
    auto blks = [](long n) { return (int)((n + 255) / 256); };
    const int NH = N * H;
    const long EF4 = (long)E * (H / 4);
    const long gemmThreads = (long)(N / 16) * (H / 16) * 32;

    // ---- degrees (with self loop) -> dinv = rsqrt(deg) ----
    k_fill<<<blks(N), TB, 0, stream>>>(dinv, 1.0f, N);
    k_count_deg<<<blks(E), TB, 0, stream>>>(dstI, dinv, E);
    k_rsqrt_inplace<<<blks(N), TB, 0, stream>>>(dinv, N);

    // ---- gcn1: xw = x @ W1 ; aggregate ; bn0+relu -> hbuf ----
    k_f32_to_f16<<<blks(N * FIN), TB, 0, stream>>>(x, a16, N * FIN);
    k_transpose_to_f16<<<blks(FIN * H), TB, 0, stream>>>(gcn1_W, w16, FIN, H);
    k_wmma_gemm<<<blks(gemmThreads), TB, 0, stream>>>(a16, w16, nullptr, xw, N, H, FIN);
    k_self_norm<<<blks(NH), TB, 0, stream>>>(xw, dinv, gcn1_b, agg, N, H);
    k_edge_scatter_norm<<<blks(EF4), TB, 0, stream>>>(xw, dinv, srcI, dstI, agg, E, H);
    k_bn_stats<<<H, TB, 0, stream>>>(agg, bnm, bnv, N, H);
    k_bn_relu<<<blks(NH), TB, 0, stream>>>(agg, bnm, bnv, bn_g[0], bn_b[0], hbuf, N, H);

    // ---- GAT stack ----
    for (int i = 0; i < 3; ++i) {
        k_f32_to_f16<<<blks(NH), TB, 0, stream>>>(hbuf, a16, NH);
        k_transpose_to_f16<<<blks(H * H), TB, 0, stream>>>(gat_W[i], w16, H, H);
        k_wmma_gemm<<<blks(gemmThreads), TB, 0, stream>>>(a16, w16, nullptr, xw, N, H, H);
        k_gat_dots<<<N, TB, 0, stream>>>(xw, gat_as[i], gat_ad[i], als, ald, N, H);
        k_init_maxsum<<<blks(N), TB, 0, stream>>>(nmax, nsum, N);
        k_gat_max<<<blks(E + N), TB, 0, stream>>>(srcI, dstI, als, ald, nmax, E, N);
        k_gat_sum<<<blks(E + N), TB, 0, stream>>>(srcI, dstI, als, ald, nmax, nsum, E, N);
        k_gat_self_out<<<blks(NH), TB, 0, stream>>>(xw, als, ald, nmax, nsum, gat_bb[i], agg, N, H);
        k_gat_edge_scatter<<<blks(EF4), TB, 0, stream>>>(xw, srcI, dstI, als, ald, nmax, nsum, agg, E, H);
        if (i > 0) {  // residual: + norm_adj(hbuf)
            k_self_norm<<<blks(NH), TB, 0, stream>>>(hbuf, dinv, nullptr, res, N, H);
            k_edge_scatter_norm<<<blks(EF4), TB, 0, stream>>>(hbuf, dinv, srcI, dstI, res, E, H);
            k_add<<<blks(NH), TB, 0, stream>>>(agg, res, agg, NH);
        }
        k_bn_stats<<<H, TB, 0, stream>>>(agg, bnm, bnv, N, H);
        k_bn_relu<<<blks(NH), TB, 0, stream>>>(agg, bnm, bnv, bn_g[i + 1], bn_b[i + 1], hbuf, N, H);
    }

    // ---- gcn2 + norm_adj residual -> agg (final node features) ----
    k_f32_to_f16<<<blks(NH), TB, 0, stream>>>(hbuf, a16, NH);
    k_transpose_to_f16<<<blks(H * H), TB, 0, stream>>>(gcn2_W, w16, H, H);
    k_wmma_gemm<<<blks(gemmThreads), TB, 0, stream>>>(a16, w16, nullptr, xw, N, H, H);
    k_self_norm<<<blks(NH), TB, 0, stream>>>(xw, dinv, gcn2_b, agg, N, H);
    k_edge_scatter_norm<<<blks(EF4), TB, 0, stream>>>(xw, dinv, srcI, dstI, agg, E, H);
    k_self_norm<<<blks(NH), TB, 0, stream>>>(hbuf, dinv, nullptr, res, N, H);
    k_edge_scatter_norm<<<blks(EF4), TB, 0, stream>>>(hbuf, dinv, srcI, dstI, res, E, H);
    k_add<<<blks(NH), TB, 0, stream>>>(agg, res, agg, NH);

    // ---- rank head: u = h@W1_top ; v = h@W1_bot + b1 ; LN stats ; fused out
    k_f32_to_f16<<<blks(NH), TB, 0, stream>>>(agg, a16, NH);
    k_transpose_to_f16<<<blks(H * H), TB, 0, stream>>>(rank_W1, w16, H, H);
    k_wmma_gemm<<<blks(gemmThreads), TB, 0, stream>>>(a16, w16, nullptr, ub, N, H, H);
    k_transpose_to_f16<<<blks(H * H), TB, 0, stream>>>(rank_W1 + (size_t)H * H, w16, H, H);
    k_wmma_gemm<<<blks(gemmThreads), TB, 0, stream>>>(a16, w16, rank_b1, vb, N, H, H);
    k_fill<<<1, TB, 0, stream>>>(Sst, 0.f, 2);
    k_rank_stats<<<blks(G * H), TB, 0, stream>>>(ub, vb, Sst, G, P, H);
    k_rank_ms<<<1, 1, 0, stream>>>(Sst, ms, Ttot);
    k_rank_out<<<blks((long)G * PAIRS * 32), TB, 0, stream>>>(
        ub, vb, ms, ln_g, ln_b, rank_W2, rank_b2, (float*)d_out, G, P, H, PAIRS);
}
